// LSTMCell_48481590837364
// MI455X (gfx1250) — matrix operand order repaired and checked by reading, not compiled, not run on hardware
//
#include <hip/hip_runtime.h>
#include <stdint.h>

// ---------------------------------------------------------------------------
// LSTM cell for MI455X (gfx1250), compile-only tuned.
//   g = x@Wx^T + h@Wh^T + bx + bh   (B=4096, H=1024, 4 gates)
//   c' = sig(gf)*c + sig(gi)*tanh(gg);  h' = sig(go)*tanh(c')
// f32->bf16 once, v_wmma_f32_16x16x32_bf16 with f32 accum, fused epilogue.
// Async global->LDS double-buffered pipeline, k-loop unrolled x2 so the
// accumulators never rotate registers; B frags grouped ahead of WMMAs.
// ---------------------------------------------------------------------------

typedef __attribute__((ext_vector_type(16))) __bf16 v16bf;
typedef __attribute__((ext_vector_type(8)))  __bf16 v8bf;
typedef __attribute__((ext_vector_type(4)))  __bf16 v4bf;
typedef __attribute__((ext_vector_type(8)))  float  v8f;
typedef int v4i_vs __attribute__((vector_size(16)));   // matches builtin param

#define ASG __attribute__((address_space(1)))
#define ASL __attribute__((address_space(3)))

#ifndef __has_builtin
#define __has_builtin(x) 0
#endif

#if defined(__HIP_DEVICE_COMPILE__) && __has_builtin(__builtin_amdgcn_global_load_async_to_lds_b128)
#define USE_ASYNC_LDS 1
#else
#define USE_ASYNC_LDS 0
#endif

#if USE_ASYNC_LDS
#if __has_builtin(__builtin_amdgcn_s_wait_asynccnt)
#define WAIT_ASYNC(N) __builtin_amdgcn_s_wait_asynccnt(N)
#else
#define WAIT_ASYNC(N) asm volatile("s_wait_asynccnt " #N ::: "memory")
#endif
#else
#define WAIT_ASYNC(N) do {} while (0)
#endif

#define TO_AS1_V4I(p) ((ASG v4i_vs*)(uintptr_t)(p))
#define TO_AS3_V4I(p) ((ASL v4i_vs*)(unsigned int)(uintptr_t)(p))

#define BATCH 4096
#define HID   1024
#define KIN   1024

#define BM 64          // batch rows per block
#define BN 64          // hidden cols per block (x4 gates in-register)
#define KT 32          // K per WMMA step
#define NKSTEP 64      // (KIN + HID) / KT

// LDS per buffer: A 64 rows x (64B data + 16B pad); B 256 rows (4g x 64c) x 80B.
#define A_ROW_B   80
#define B_ROW_B   80
#define A_TILE_B  (BM * A_ROW_B)                // 5120
#define B_TILE_B  (4 * BN * B_ROW_B)            // 20480
#define BUF_B     (A_TILE_B + B_TILE_B)         // 25600
#define SMEM_B    (2 * BUF_B)                   // 51200

// ---------------------------------------------------------------------------
// Pass 1: convert x, h, Wx, Wh (each 4M f32) to bf16 in workspace.
// ---------------------------------------------------------------------------
__device__ __forceinline__ v4bf cvt4(float4 f) {
    v4bf r;
    r.x = (__bf16)f.x; r.y = (__bf16)f.y; r.z = (__bf16)f.z; r.w = (__bf16)f.w;
    return r;
}

__global__ __launch_bounds__(256) void lstm_cvt_bf16(
    const float4* __restrict__ x,  const float4* __restrict__ h,
    const float4* __restrict__ wx, const float4* __restrict__ wh,
    v4bf* __restrict__ xb, v4bf* __restrict__ hb,
    v4bf* __restrict__ wxb, v4bf* __restrict__ whb) {
    int i = blockIdx.x * 256 + threadIdx.x;      // 1M float4 groups per array
    xb[i]  = cvt4(x[i]);
    hb[i]  = cvt4(h[i]);
    wxb[i] = cvt4(wx[i]);
    whb[i] = cvt4(wh[i]);
}

// ---------------------------------------------------------------------------
// 16-byte global -> LDS copy (async path verified on this toolchain)
// ---------------------------------------------------------------------------
__device__ __forceinline__ void cp16_g2l(const __bf16* __restrict__ g,
                                         unsigned char* l) {
#if USE_ASYNC_LDS
    __builtin_amdgcn_global_load_async_to_lds_b128(
        TO_AS1_V4I(g), TO_AS3_V4I(l), 0, 0);
#else
    v8bf v = *(const v8bf*)g;
    *(v8bf*)l = v;
#endif
}

__device__ __forceinline__ float sigf(float x) {
    return 1.0f / (1.0f + __expf(-x));
}
__device__ __forceinline__ float tanh_fast(float x) {
    return 2.0f / (1.0f + __expf(-2.0f * x)) - 1.0f;   // 2*sig(2x)-1
}

// ---------------------------------------------------------------------------
// Pass 2: fused GEMM (x@Wx^T + h@Wh^T) + LSTM epilogue.
// grid = (BATCH/BM, HID/BN); block = 256 (8 waves).
// wave w: row tile rt = w&3 (16 rows), col half cw = w>>2 (32 cols = 2 subtiles),
// 4 gates in-register -> 8 accumulator fragments per wave.
// ---------------------------------------------------------------------------
__global__ __launch_bounds__(256) void lstm_wmma_fused(
    const __bf16* __restrict__ xb,  const __bf16* __restrict__ hb,
    const __bf16* __restrict__ wxb, const __bf16* __restrict__ whb,
    const float* __restrict__ cin,
    const float* __restrict__ bx,   const float* __restrict__ bh,
    float* __restrict__ out_h, float* __restrict__ out_c) {

    __shared__ __align__(16) unsigned char smem[SMEM_B];

    const int tid  = threadIdx.x;
    const int wave = tid >> 5;
    const int lane = tid & 31;
    const int rt   = wave & 3;          // 0..3 (16-row tile)
    const int cw   = wave >> 2;         // 0..1 (32-col half)
    const int r0   = blockIdx.x * BM;   // batch base
    const int c0   = blockIdx.y * BN;   // hidden base

    // --- tile stager: 256 lanes move A (256x16B) + B (1024x16B) ------------
    auto stage = [&](int ks, int buf) {
        const __bf16* Asrc; const __bf16* Wsrc; int k0;
        if (ks < 32) { Asrc = xb; Wsrc = wxb; k0 = ks * KT; }
        else         { Asrc = hb; Wsrc = whb; k0 = (ks - 32) * KT; }
        unsigned char* base = smem + buf * BUF_B;
        {   // A: 64 rows x 32 k (thread -> row = tid>>2, 8-elem seg = tid&3)
            int row = tid >> 2, seg = tid & 3;
            cp16_g2l(Asrc + (size_t)(r0 + row) * KIN + k0 + seg * 8,
                     base + row * A_ROW_B + seg * 16);
        }
#pragma unroll
        for (int t = 0; t < 4; ++t) {   // B: 4 gates x 64 cols x 4 segs
            int idx = tid + 256 * t;
            int seg = idx & 3, cc = (idx >> 2) & 63, g = idx >> 8;
            cp16_g2l(Wsrc + (size_t)(g * HID + c0 + cc) * KIN + k0 + seg * 8,
                     base + A_TILE_B + (g * BN + cc) * B_ROW_B + seg * 16);
        }
    };

    v8f acc[4][2] = {};

    // --- one k-step of math from LDS buffer `buf` ---------------------------
    auto compute = [&](int buf) {
        const unsigned char* base = smem + buf * BUF_B;
        // A fragment (16x32): lane<16 K{0..7,16..23}, lane>=16 K{8..15,24..31}
        const unsigned char* arow =
            base + (rt * 16 + (lane & 15)) * A_ROW_B + ((lane >> 4) * 16);
        v8bf alo = *(const v8bf*)(arow);
        v8bf ahi = *(const v8bf*)(arow + 32);
        v16bf afrag = __builtin_shufflevector(alo, ahi,
            0, 1, 2, 3, 4, 5, 6, 7, 8, 9, 10, 11, 12, 13, 14, 15);

#pragma unroll
        for (int gp = 0; gp < 2; ++gp) {        // gate pairs: {0,1}, {2,3}
            v16bf bfrag[2][2];
#pragma unroll
            for (int gi = 0; gi < 2; ++gi) {
#pragma unroll
                for (int j = 0; j < 2; ++j) {
                    const unsigned char* brow = base + A_TILE_B +
                        ((gp * 2 + gi) * BN + cw * 32 + j * 16 + (lane & 15)) * B_ROW_B +
                        (lane >> 4) * 16;
                    v8bf blo = *(const v8bf*)(brow);
                    v8bf bhi = *(const v8bf*)(brow + 32);
                    bfrag[gi][j] = __builtin_shufflevector(blo, bhi,
                        0, 1, 2, 3, 4, 5, 6, 7, 8, 9, 10, 11, 12, 13, 14, 15);
                }
            }
#pragma unroll
            for (int gi = 0; gi < 2; ++gi)
#pragma unroll
                for (int j = 0; j < 2; ++j)
                    acc[gp * 2 + gi][j] = __builtin_amdgcn_wmma_f32_16x16x32_bf16(
                        false, afrag, false, bfrag[gi][j], (short)0,
                        acc[gp * 2 + gi][j], false, false);
        }
    };

    // --- double-buffered pipeline, k-loop unrolled x2 (no buffer rotation) --
    stage(0, 0);
    for (int ks = 0; ks < NKSTEP; ks += 2) {
        stage(ks + 1, 1);          // ks+1 <= 63 always
        WAIT_ASYNC(5);             // oldest 5 = buf0's loads
        __syncthreads();
        compute(0);
        __syncthreads();           // all waves done reading buf0
        if (ks + 2 < NKSTEP) { stage(ks + 2, 0); WAIT_ASYNC(5); }
        else                 { WAIT_ASYNC(0); }
        __syncthreads();
        compute(1);
        __syncthreads();           // all waves done reading buf1
    }

    // --- fused LSTM epilogue ------------------------------------------------
#pragma unroll
    for (int j = 0; j < 2; ++j) {
        const int col = c0 + cw * 32 + j * 16 + (lane & 15);
        const float bI = bx[0 * HID + col] + bh[0 * HID + col];
        const float bF = bx[1 * HID + col] + bh[1 * HID + col];
        const float bG = bx[2 * HID + col] + bh[2 * HID + col];
        const float bO = bx[3 * HID + col] + bh[3 * HID + col];
#pragma unroll
        for (int v = 0; v < 8; ++v) {
            const int row = r0 + rt * 16 + v + 8 * (lane >> 4);
            const size_t off = (size_t)row * HID + col;
            const float gi = sigf(acc[0][j][v] + bI);
            const float gf = sigf(acc[1][j][v] + bF);
            const float gg = tanh_fast(acc[2][j][v] + bG);
            const float go = sigf(acc[3][j][v] + bO);
            const float cn = gf * cin[off] + gi * gg;
            out_c[off] = cn;
            out_h[off] = go * tanh_fast(cn);
        }
    }
}

// ---------------------------------------------------------------------------
extern "C" void kernel_launch(void* const* d_in, const int* in_sizes, int n_in,
                              void* d_out, int out_size, void* d_ws, size_t ws_size,
                              hipStream_t stream) {
    (void)in_sizes; (void)n_in; (void)out_size; (void)ws_size;

    const float* x  = (const float*)d_in[0];
    const float* h  = (const float*)d_in[1];
    const float* c  = (const float*)d_in[2];
    const float* Wx = (const float*)d_in[3];
    const float* Wh = (const float*)d_in[4];
    const float* bx = (const float*)d_in[5];
    const float* bh = (const float*)d_in[6];
    float* out = (float*)d_out;

    const size_t N4M = (size_t)4096 * 1024;   // elements in each of x,h,Wx,Wh
    __bf16* xb  = (__bf16*)d_ws;
    __bf16* hbf = xb  + N4M;
    __bf16* wxb = hbf + N4M;
    __bf16* whb = wxb + N4M;

    // Pass 1: f32 -> bf16 (1M float4 groups per array)
    lstm_cvt_bf16<<<4096, 256, 0, stream>>>(
        (const float4*)x, (const float4*)h, (const float4*)Wx, (const float4*)Wh,
        (v4bf*)xb, (v4bf*)hbf, (v4bf*)wxb, (v4bf*)whb);

    // Pass 2: fused WMMA GEMM + LSTM epilogue
    dim3 grid(BATCH / BM, HID / BN);
    lstm_wmma_fused<<<grid, 256, 0, stream>>>(
        xb, hbf, wxb, whb, c, bx, bh, out, out + (size_t)BATCH * HID);
}
